// GAT_63582695850892
// MI455X (gfx1250) — matrix operand order repaired and checked by reading, not compiled, not run on hardware
//
#include <hip/hip_runtime.h>
#include <hip/hip_bf16.h>
#include <stdint.h>

typedef __bf16 bf16;
typedef __attribute__((ext_vector_type(16))) __bf16 v16bf;
typedef __attribute__((ext_vector_type(8)))  __bf16 v8bf;
typedef __attribute__((ext_vector_type(8)))  float  v8f;

// ---- order-preserving float<->uint encoding for atomic max -----------------
__device__ __forceinline__ unsigned gat_enc(float f) {
  unsigned u = __float_as_uint(f);
  return (u & 0x80000000u) ? ~u : (u | 0x80000000u);
}
__device__ __forceinline__ float gat_dec(unsigned u) {
  return (u & 0x80000000u) ? __uint_as_float(u & 0x7fffffffu)
                           : __uint_as_float(~u);
}

// ---- elementwise helpers ---------------------------------------------------
__global__ void gat_f32_to_bf16(const float* __restrict__ src,
                                bf16* __restrict__ dst, int n) {
  int i = blockIdx.x * blockDim.x + threadIdx.x;
  if (i < n) dst[i] = (bf16)src[i];
}

__global__ void gat_zero_u32(unsigned* __restrict__ p, int n) {
  int i = blockIdx.x * blockDim.x + threadIdx.x;
  if (i < n) p[i] = 0u;
}

// ---- WMMA GEMM: C[M,N] = A[M,K] * W[N,K]^T  (bf16 in, f32 out) -------------
// One wave (32 lanes) per 16x16 C tile; K-loop step 32 using
// v_wmma_f32_16x16x32_bf16.
__global__ __launch_bounds__(32) void gat_wmma_gemm(
    const bf16* __restrict__ A, const bf16* __restrict__ W,
    float* __restrict__ C, int M, int K, int N)
{
  const int lane = threadIdx.x;
  const int g = lane >> 4;          // lane group 0/1
  const int r = lane & 15;
  const int tm = blockIdx.x, tn = blockIdx.y;

  int mA = tm * 16 + r; if (mA >= M) mA = M - 1;   // branchless clamp
  const bf16* arow = A + (size_t)mA * K;
  const bf16* wrow = W + (size_t)(tn * 16 + r) * K;

  v8f acc = {};
  for (int k0 = 0; k0 < K; k0 += 32) {
    // A 16x32 bf16 layout: group g holds K = g*8..g*8+7 (v0..3) and
    // 16+g*8..16+g*8+7 (v4..7), row m = lane%16.
    v8bf a0 = *(const v8bf*)(arow + k0 + g * 8);
    v8bf a1 = *(const v8bf*)(arow + k0 + 16 + g * 8);
    v16bf a;
#pragma unroll
    for (int i = 0; i < 8; ++i) { a[i] = a0[i]; a[8 + i] = a1[i]; }
    // B 32x16 bf16: column n = lane%16, group g holds K = g*8..g*8+7 and 16+g*8..16+g*8+7.
    v8bf b0 = *(const v8bf*)(wrow + k0 + g * 8);
    v8bf b1 = *(const v8bf*)(wrow + k0 + 16 + g * 8);
    v16bf b;
#pragma unroll
    for (int i = 0; i < 8; ++i) { b[i] = b0[i]; b[8 + i] = b1[i]; }
    acc = __builtin_amdgcn_wmma_f32_16x16x32_bf16(
        false, a, false, b, (short)0, acc, false, false);
  }
  // C/D: VGPR i -> row tm*16 + g*8 + i, column tn*16 + lane%16
  const int row0 = tm * 16 + g * 8;
  float* cp = C + (size_t)row0 * N + tn * 16 + r;
#pragma unroll
  for (int i = 0; i < 8; ++i)
    if (row0 + i < M) cp[(size_t)i * N] = acc[i];
}

// ---- per-(node, head) attention scores: a_src, a_dst -----------------------
template<int H, int C>
__global__ void gat_att_scores(const float* __restrict__ xw,
                               const float* __restrict__ att_s,
                               const float* __restrict__ att_d,
                               float* __restrict__ as_, float* __restrict__ ad_,
                               int n)
{
  int idx = blockIdx.x * blockDim.x + threadIdx.x;
  if (idx >= n * H) return;
  int node = idx / H, h = idx % H;
  const float* v = xw + (size_t)node * (H * C) + h * C;
  float s = 0.f, d = 0.f;
#pragma unroll
  for (int c = 0; c < C; ++c) {
    float x = v[c];
    s = fmaf(x, att_s[h * C + c], s);
    d = fmaf(x, att_d[h * C + c], d);
  }
  as_[idx] = s; ad_[idx] = d;
}

// ---- edge helpers ----------------------------------------------------------
__device__ __forceinline__ void gat_edge_ends(const int* ei, int Eraw, int e,
                                              int& src, int& dst) {
  if (e < Eraw) { src = ei[e]; dst = ei[Eraw + e]; }
  else          { src = e - Eraw; dst = e - Eraw; }     // self loops
}
__device__ __forceinline__ float gat_lrelu(float a) {
  return a > 0.f ? a : 0.2f * a;
}

// pass 1: per-destination max of leaky_relu(alpha)
template<int H>
__global__ void gat_edge_max(const int* __restrict__ ei, int Eraw, int Etot,
                             const float* __restrict__ as_,
                             const float* __restrict__ ad_,
                             unsigned* __restrict__ mx)
{
  long long idx = blockIdx.x * (long long)blockDim.x + threadIdx.x;
  if (idx >= (long long)Etot * H) return;
  int e = (int)(idx / H), h = (int)(idx % H);
  int src, dst; gat_edge_ends(ei, Eraw, e, src, dst);
  float a = gat_lrelu(as_[src * H + h] + ad_[dst * H + h]);
  atomicMax(mx + dst * H + h, gat_enc(a));
}

// pass 2: per-destination sum of exp(alpha - max)
template<int H>
__global__ void gat_edge_sum(const int* __restrict__ ei, int Eraw, int Etot,
                             const float* __restrict__ as_,
                             const float* __restrict__ ad_,
                             const unsigned* __restrict__ mx,
                             float* __restrict__ sm)
{
  long long idx = blockIdx.x * (long long)blockDim.x + threadIdx.x;
  if (idx >= (long long)Etot * H) return;
  int e = (int)(idx / H), h = (int)(idx % H);
  int src, dst; gat_edge_ends(ei, Eraw, e, src, dst);
  float a = gat_lrelu(as_[src * H + h] + ad_[dst * H + h]);
  unsafeAtomicAdd(sm + dst * H + h, expf(a - gat_dec(mx[dst * H + h])));
}

// pass 3: agg[dst] += softmax_coef * xw[src]   (one thread per edge-channel)
template<int H, int C>
__global__ void gat_edge_agg(const int* __restrict__ ei, int Eraw, int Etot,
                             const float* __restrict__ as_,
                             const float* __restrict__ ad_,
                             const unsigned* __restrict__ mx,
                             const float* __restrict__ sm,
                             const float* __restrict__ xw,
                             float* __restrict__ agg)
{
  const int CT = H * C;
  long long idx = blockIdx.x * (long long)blockDim.x + threadIdx.x;
  if (idx >= (long long)Etot * CT) return;
  int e = (int)(idx / CT), c = (int)(idx % CT), h = c / C;
  int src, dst; gat_edge_ends(ei, Eraw, e, src, dst);
  float a = gat_lrelu(as_[src * H + h] + ad_[dst * H + h]);
  float coef = expf(a - gat_dec(mx[dst * H + h])) / sm[dst * H + h];
  unsafeAtomicAdd(&agg[(size_t)dst * CT + c],
                  coef * xw[(size_t)src * CT + c]);
}

// ---- layer-1 epilogue: h = elu(agg + bias1) + skip1 + skip_b1 (in place),
//      plus bf16 copy for layer-2 WMMA GEMMs --------------------------------
__global__ void gat_final1(const float* __restrict__ agg,
                           const float* __restrict__ bias,
                           float* skip_h,                     // read skip, write h
                           const float* __restrict__ skip_b,
                           bf16* __restrict__ hbf, int total, int CT)
{
  int i = blockIdx.x * blockDim.x + threadIdx.x;
  if (i >= total) return;
  int c = i % CT;
  float v = agg[i] + bias[c];
  v = v > 0.f ? v : expm1f(v);                // ELU, alpha=1
  v += skip_h[i] + skip_b[c];
  skip_h[i] = v;
  hbf[i] = (bf16)v;
}

// ---- layer-2 epilogue: out = agg2 + bias2 + skip2 + skip_b2 ----------------
__global__ void gat_final2(const float* __restrict__ agg,
                           const float* __restrict__ bias,
                           const float* __restrict__ skip,
                           const float* __restrict__ skip_b,
                           float* __restrict__ out, int total, int CT)
{
  int i = blockIdx.x * blockDim.x + threadIdx.x;
  if (i >= total) return;
  int c = i % CT;
  out[i] = agg[i] + bias[c] + skip[i] + skip_b[c];
}

// ---------------------------------------------------------------------------
extern "C" void kernel_launch(void* const* d_in, const int* in_sizes, int n_in,
                              void* d_out, int out_size, void* d_ws, size_t ws_size,
                              hipStream_t stream)
{
  const float* x       = (const float*)d_in[0];
  const int*   ei      = (const int*)  d_in[1];
  const float* lin_w1  = (const float*)d_in[2];
  const float* att_s1  = (const float*)d_in[3];
  const float* att_d1  = (const float*)d_in[4];
  const float* bias1   = (const float*)d_in[5];
  const float* lin_w2  = (const float*)d_in[6];
  const float* att_s2  = (const float*)d_in[7];
  const float* att_d2  = (const float*)d_in[8];
  const float* bias2   = (const float*)d_in[9];
  const float* skip_w1 = (const float*)d_in[10];
  const float* skip_b1 = (const float*)d_in[11];
  const float* skip_w2 = (const float*)d_in[12];
  const float* skip_b2 = (const float*)d_in[13];

  const int Nn   = in_sizes[0] / 128;   // 50000 nodes
  const int Eraw = in_sizes[1] / 2;     // 800000 edges
  const int Etot = Eraw + Nn;           // + self loops
  (void)n_in; (void)out_size; (void)ws_size;

  // ---- workspace bump allocator (256B aligned) ----
  char* base = (char*)d_ws; size_t off = 0;
  auto alloc = [&](size_t bytes) -> void* {
    off = (off + 255) & ~(size_t)255;
    void* p = base + off; off += bytes; return p;
  };
  float*    xw1   = (float*)   alloc((size_t)Nn * 256 * 4);
  float*    agg1  = (float*)   alloc((size_t)Nn * 256 * 4);
  float*    skiph = (float*)   alloc((size_t)Nn * 256 * 4);  // skip1, then h
  float*    xw2   = (float*)   alloc((size_t)Nn * 64 * 4);
  float*    skip2 = (float*)   alloc((size_t)Nn * 64 * 4);
  float*    agg2  = (float*)   alloc((size_t)Nn * 64 * 4);
  float*    as1   = (float*)   alloc((size_t)Nn * 8 * 4);
  float*    ad1   = (float*)   alloc((size_t)Nn * 8 * 4);
  unsigned* m1    = (unsigned*)alloc((size_t)Nn * 8 * 4);
  float*    s1    = (float*)   alloc((size_t)Nn * 8 * 4);
  float*    as2   = (float*)   alloc((size_t)Nn * 4);
  float*    ad2   = (float*)   alloc((size_t)Nn * 4);
  unsigned* m2    = (unsigned*)alloc((size_t)Nn * 4);
  float*    s2    = (float*)   alloc((size_t)Nn * 4);
  bf16*     xbf   = (bf16*)    alloc((size_t)Nn * 128 * 2);
  bf16*     hbf   = (bf16*)    alloc((size_t)Nn * 256 * 2);
  bf16*     w1b   = (bf16*)    alloc((size_t)256 * 128 * 2);
  bf16*     sw1b  = (bf16*)    alloc((size_t)256 * 128 * 2);
  bf16*     w2b   = (bf16*)    alloc((size_t)64 * 256 * 2);
  bf16*     sw2b  = (bf16*)    alloc((size_t)64 * 256 * 2);

  auto cdiv = [](long long a, long long b) { return (unsigned)((a + b - 1) / b); };
  const int TB = 256;

  // ---- convert weights + x to bf16 ----
  gat_f32_to_bf16<<<cdiv(256 * 128, TB), TB, 0, stream>>>(lin_w1,  w1b,  256 * 128);
  gat_f32_to_bf16<<<cdiv(256 * 128, TB), TB, 0, stream>>>(skip_w1, sw1b, 256 * 128);
  gat_f32_to_bf16<<<cdiv(64 * 256, TB),  TB, 0, stream>>>(lin_w2,  w2b,  64 * 256);
  gat_f32_to_bf16<<<cdiv(64 * 256, TB),  TB, 0, stream>>>(skip_w2, sw2b, 64 * 256);
  gat_f32_to_bf16<<<cdiv((long long)Nn * 128, TB), TB, 0, stream>>>(x, xbf, Nn * 128);

  // ---- layer 1: WMMA GEMMs (xw1, skip1) ----
  dim3 g1(cdiv(Nn, 16), 256 / 16);
  gat_wmma_gemm<<<g1, 32, 0, stream>>>(xbf, w1b,  xw1,   Nn, 128, 256);
  gat_wmma_gemm<<<g1, 32, 0, stream>>>(xbf, sw1b, skiph, Nn, 128, 256);

  gat_att_scores<8, 32><<<cdiv(Nn * 8, TB), TB, 0, stream>>>(xw1, att_s1, att_d1, as1, ad1, Nn);
  gat_zero_u32<<<cdiv(Nn * 8, TB), TB, 0, stream>>>(m1, Nn * 8);
  gat_zero_u32<<<cdiv(Nn * 8, TB), TB, 0, stream>>>((unsigned*)s1, Nn * 8);
  gat_zero_u32<<<cdiv((long long)Nn * 256, TB), TB, 0, stream>>>((unsigned*)agg1, Nn * 256);

  gat_edge_max<8><<<cdiv((long long)Etot * 8, TB), TB, 0, stream>>>(ei, Eraw, Etot, as1, ad1, m1);
  gat_edge_sum<8><<<cdiv((long long)Etot * 8, TB), TB, 0, stream>>>(ei, Eraw, Etot, as1, ad1, m1, s1);
  gat_edge_agg<8, 32><<<cdiv((long long)Etot * 256, TB), TB, 0, stream>>>(
      ei, Eraw, Etot, as1, ad1, m1, s1, xw1, agg1);

  gat_final1<<<cdiv((long long)Nn * 256, TB), TB, 0, stream>>>(
      agg1, bias1, skiph, skip_b1, hbf, Nn * 256, 256);

  // ---- layer 2: WMMA GEMMs (xw2, skip2) ----
  dim3 g2(cdiv(Nn, 16), 64 / 16);
  gat_wmma_gemm<<<g2, 32, 0, stream>>>(hbf, w2b,  xw2,   Nn, 256, 64);
  gat_wmma_gemm<<<g2, 32, 0, stream>>>(hbf, sw2b, skip2, Nn, 256, 64);

  gat_att_scores<1, 64><<<cdiv(Nn, TB), TB, 0, stream>>>(xw2, att_s2, att_d2, as2, ad2, Nn);
  gat_zero_u32<<<cdiv(Nn, TB), TB, 0, stream>>>(m2, Nn);
  gat_zero_u32<<<cdiv(Nn, TB), TB, 0, stream>>>((unsigned*)s2, Nn);
  gat_zero_u32<<<cdiv((long long)Nn * 64, TB), TB, 0, stream>>>((unsigned*)agg2, Nn * 64);

  gat_edge_max<1><<<cdiv(Etot, TB), TB, 0, stream>>>(ei, Eraw, Etot, as2, ad2, m2);
  gat_edge_sum<1><<<cdiv(Etot, TB), TB, 0, stream>>>(ei, Eraw, Etot, as2, ad2, m2, s2);
  gat_edge_agg<1, 64><<<cdiv((long long)Etot * 64, TB), TB, 0, stream>>>(
      ei, Eraw, Etot, as2, ad2, m2, s2, xw2, agg2);

  gat_final2<<<cdiv((long long)Nn * 64, TB), TB, 0, stream>>>(
      agg2, bias2, skip2, skip_b2, (float*)d_out, Nn * 64, 64);
}
